// StageBlock_14774687498735
// MI455X (gfx1250) — compile-verified, matrix operand order checked by reading, not run on hardware
//
#include <hip/hip_runtime.h>
#include <hip/hip_bf16.h>

// ---------------- types for CDNA5 WMMA ----------------
typedef __attribute__((ext_vector_type(16))) __bf16 v16bf;
typedef __attribute__((ext_vector_type(8)))  __bf16 v8bf;
typedef __attribute__((ext_vector_type(8)))  float  v8f;

#define KDIM 768
#define NH 6
#define NL 3
#define NP 4
#define DH 128

// Load a 16x32 bf16 WMMA fragment for this lane.
// Per CDNA5 ISA: lanes 0-15 hold K = 0-7 and 16-23; lanes 16-31 hold K = 8-15 and 24-31
// (the +kb base is folded into the pointer by the caller). Two contiguous 16B chunks
// at +0 and +32 bytes.
__device__ __forceinline__ v16bf load_frag(const __bf16* p) {
    v8bf lo = *(const v8bf*)(p);
    v8bf hi = *(const v8bf*)(p + 16);
    v16bf r;
#pragma unroll
    for (int i = 0; i < 8; ++i) { r[i] = lo[i]; r[i + 8] = hi[i]; }
    return r;
}

// ---------------- weight transpose + cast: W[768][N] f32 -> WT[Npad][768] bf16 ----------------
__global__ __launch_bounds__(256) void wtrans_kernel(const float* __restrict__ W,
                                                     __bf16* __restrict__ WT,
                                                     int N, int Npad) {
    int t = blockIdx.x * 256 + threadIdx.x;
    if (t >= Npad * KDIM) return;
    int n = t / KDIM;
    int k = t - n * KDIM;
    WT[t] = (n < N) ? (__bf16)W[(size_t)k * N + n] : (__bf16)0.0f;
}

// ---------------- LayerNorm (row of 768) -> bf16 ----------------
__global__ __launch_bounds__(256) void ln_kernel(const float* __restrict__ x,
                                                 const float* __restrict__ g,
                                                 const float* __restrict__ b,
                                                 __bf16* __restrict__ out) {
    int row = blockIdx.x;
    int tid = threadIdx.x;
    const float* xr = x + (size_t)row * KDIM;
    float v0 = xr[tid], v1 = xr[tid + 256], v2 = xr[tid + 512];
    __shared__ float sh1[256], sh2[256];
    sh1[tid] = v0 + v1 + v2;
    sh2[tid] = v0 * v0 + v1 * v1 + v2 * v2;
    __syncthreads();
    for (int off = 128; off > 0; off >>= 1) {
        if (tid < off) { sh1[tid] += sh1[tid + off]; sh2[tid] += sh2[tid + off]; }
        __syncthreads();
    }
    float mu  = sh1[0] * (1.0f / 768.0f);
    float var = sh2[0] * (1.0f / 768.0f) - mu * mu;
    float rs  = rsqrtf(var + 1e-6f);
    __bf16* orow = out + (size_t)row * KDIM;
    orow[tid]       = (__bf16)((v0 - mu) * rs * g[tid]       + b[tid]);
    orow[tid + 256] = (__bf16)((v1 - mu) * rs * g[tid + 256] + b[tid + 256]);
    orow[tid + 512] = (__bf16)((v2 - mu) * rs * g[tid + 512] + b[tid + 512]);
}

// ---------------- bf16 WMMA GEMM with async-to-LDS B staging ----------------
// Block: 256 threads = 8 waves. Block tile 128(M) x 64(N). Wave tile 16 x 64 (4 wmma accs).
// B tile (64 x 32 K, 4KB) is staged in LDS via global_load_async_to_lds_b128,
// double-buffered, synchronized with s_wait_asynccnt + workgroup barrier.
// All four B fragments are loaded from LDS before the WMMA burst so the matrix
// pipe sees back-to-back v_wmma ops behind a single dscnt wait.
// MODE 0: value proj  -> outf[((b*6+h)*5376+pos)*128+d] = acc + bias[col]
// MODE 1: plain       -> outf[row*N+col] = acc + bias[col]
// MODE 2: prompt      -> p=(acc+bias[col])*scale[col]; outf[row*768+col]=p; outb=bf16(p)
// MODE 3: residual    -> outf[row*768+col] = acc + bias[col] + addsrc[row*768+col]
template <int MODE>
__global__ __launch_bounds__(256) void gemm_wmma_kernel(
    const __bf16* __restrict__ A, const __bf16* __restrict__ BT,
    const float* __restrict__ bias, const float* __restrict__ scale,
    const float* __restrict__ addsrc, float* __restrict__ outf,
    __bf16* __restrict__ outb, int M, int N) {
    __shared__ __bf16 Bsh[2][64 * 32];   // [buf][n*32 + k], 4KB per buffer

    const int tid  = threadIdx.x;
    const int lane = tid & 31;
    const int wave = tid >> 5;
    const int l15  = lane & 15;
    const int hl   = lane >> 4;          // 0 | 1
    const int kb   = hl * 8;             // K base for this half-wave
    const int mBase = blockIdx.y * 128 + wave * 16;
    const int nBase = blockIdx.x * 64;

    // async producer mapping: one b128 per thread per K-step
    const int pn = tid >> 2;             // 0..63 : column of B (row of BT)
    const int pk = (tid & 3) * 8;        // 0,8,16,24 : K element offset
    const __bf16* pgB = BT + (size_t)(nBase + pn) * KDIM + pk;
    const unsigned ldsB0 =
        (unsigned)(uintptr_t)(&Bsh[0][pn * 32 + pk]);
    const unsigned ldsB1 =
        (unsigned)(uintptr_t)(&Bsh[1][pn * 32 + pk]);

    const __bf16* pa = A + (size_t)(mBase + l15) * KDIM + kb;

    v8f acc[4];
#pragma unroll
    for (int nt = 0; nt < 4; ++nt) acc[nt] = (v8f)0.0f;

    // preload K-step 0 into buffer 0
    asm volatile("global_load_async_to_lds_b128 %0, %1, off"
                 :: "v"(ldsB0), "v"(pgB) : "memory");

    for (int k0 = 0; k0 < KDIM; k0 += 32) {
        const int buf = (k0 >> 5) & 1;
        if (k0 + 32 < KDIM) {
            // prefetch next K-step into the other buffer, then retire oldest
            const unsigned dst = buf ? ldsB0 : ldsB1;
            asm volatile("global_load_async_to_lds_b128 %0, %1, off"
                         :: "v"(dst), "v"(pgB + k0 + 32) : "memory");
            asm volatile("s_wait_asynccnt 0x1" ::: "memory");
        } else {
            asm volatile("s_wait_asynccnt 0x0" ::: "memory");
        }
        __syncthreads();                 // whole B tile visible in LDS

        __builtin_prefetch(pa + k0 + 64, 0, 0);
        v16bf a = load_frag(pa + k0);
        const __bf16* bbase = &Bsh[buf][0];

        // batch all LDS fragment reads, then burst the WMMAs
        v16bf bfrag[4];
#pragma unroll
        for (int nt = 0; nt < 4; ++nt)
            bfrag[nt] = load_frag(bbase + (nt * 16 + l15) * 32 + kb);
#pragma unroll
        for (int nt = 0; nt < 4; ++nt)
            acc[nt] = __builtin_amdgcn_wmma_f32_16x16x32_bf16(
                false, a, false, bfrag[nt], (short)0, acc[nt], false, false);

        __syncthreads();                 // protect buffer before next overwrite
    }

#pragma unroll
    for (int nt = 0; nt < 4; ++nt) {
        int col = nBase + nt * 16 + l15;
        if (col >= N) continue;
#pragma unroll
        for (int r = 0; r < 8; ++r) {
            int row = mBase + r + hl * 8;
            float v = acc[nt][r];
            if (MODE == 0) {
                v += bias[col];
                int bb = row / 5376, pos = row - bb * 5376;
                int h = col >> 7, d = col & 127;
                outf[(((size_t)bb * NH + h) * 5376 + pos) * DH + d] = v;
            } else if (MODE == 1) {
                outf[(size_t)row * N + col] = v + bias[col];
            } else if (MODE == 2) {
                float p = (v + bias[col]) * scale[col];
                outf[(size_t)row * KDIM + col] = p;
                outb[(size_t)row * KDIM + col] = (__bf16)p;
            } else {
                outf[(size_t)row * KDIM + col] =
                    v + bias[col] + addsrc[(size_t)row * KDIM + col];
            }
        }
    }
}

// ---------------- multi-scale deformable sampling ----------------
// grid = B*Q*NH blocks, 128 threads (= dh channels)
__global__ __launch_bounds__(128) void msdeform_kernel(
    const float* __restrict__ off, const float* __restrict__ awl,
    const float* __restrict__ ref, const float* __restrict__ value,
    __bf16* __restrict__ attn_bf) {
    int bqh = blockIdx.x;
    int h   = bqh % NH;
    int row = bqh / NH;                 // b*4096 + q
    int b   = row >> 12;
    int d   = threadIdx.x;

    // softmax over L*P = 12 logits
    const float* lg = awl + (size_t)row * (NH * NL * NP) + h * (NL * NP);
    float l12[12], mx = -1e30f;
#pragma unroll
    for (int i = 0; i < 12; ++i) { l12[i] = lg[i]; mx = fmaxf(mx, l12[i]); }
    float s = 0.0f;
#pragma unroll
    for (int i = 0; i < 12; ++i) { l12[i] = __expf(l12[i] - mx); s += l12[i]; }
    float inv = 1.0f / s;

    const float* offp  = off + (size_t)row * (NH * NL * NP * 2) + h * (NL * NP * 2);
    const float* refp  = ref + (size_t)row * (NL * 2);
    const float* vbase = value + ((size_t)b * NH + h) * 5376 * DH + d;

    const int Ws[NL] = {64, 32, 16};
    const int Hs[NL] = {64, 32, 16};
    const int S0[NL] = {0, 4096, 5120};

    float acc = 0.0f;
#pragma unroll
    for (int l = 0; l < NL; ++l) {
        const int Wl = Ws[l], Hl = Hs[l], s0 = S0[l];
        const float fW = (float)Wl, fH = (float)Hl;
        const float rx = refp[l * 2 + 0], ry = refp[l * 2 + 1];
#pragma unroll
        for (int p = 0; p < NP; ++p) {
            float ox = offp[(l * NP + p) * 2 + 0];
            float oy = offp[(l * NP + p) * 2 + 1];
            float px = (rx + ox / fW) * fW - 0.5f;
            float py = (ry + oy / fH) * fH - 0.5f;
            float x0f = floorf(px), y0f = floorf(py);
            float lx = px - x0f, ly = py - y0f;
            int x0 = (int)x0f, y0 = (int)y0f;
            int x1 = x0 + 1, y1 = y0 + 1;
            float w00 = (1.0f - lx) * (1.0f - ly);
            float w10 = lx * (1.0f - ly);
            float w01 = (1.0f - lx) * ly;
            float w11 = lx * ly;
            float aww = l12[l * NP + p] * inv;
            float sacc = 0.0f;
            if (x0 >= 0 && x0 < Wl && y0 >= 0 && y0 < Hl)
                sacc += w00 * vbase[(size_t)(s0 + y0 * Wl + x0) * DH];
            if (x1 >= 0 && x1 < Wl && y0 >= 0 && y0 < Hl)
                sacc += w10 * vbase[(size_t)(s0 + y0 * Wl + x1) * DH];
            if (x0 >= 0 && x0 < Wl && y1 >= 0 && y1 < Hl)
                sacc += w01 * vbase[(size_t)(s0 + y1 * Wl + x0) * DH];
            if (x1 >= 0 && x1 < Wl && y1 >= 0 && y1 < Hl)
                sacc += w11 * vbase[(size_t)(s0 + y1 * Wl + x1) * DH];
            acc += aww * sacc;
        }
    }
    attn_bf[(size_t)row * KDIM + h * DH + d] = (__bf16)acc;
}

// ---------------- host launch ----------------
extern "C" void kernel_launch(void* const* d_in, const int* in_sizes, int n_in,
                              void* d_out, int out_size, void* d_ws, size_t ws_size,
                              hipStream_t stream) {
    (void)in_sizes; (void)n_in; (void)out_size; (void)ws_size;
    const int B = 4, Q = 4096, C = KDIM, Lin = 5376;
    const int MQ = B * Q;        // 16384
    const int MV = B * Lin;      // 21504
    const int NO = NH * NL * NP * 2;  // 144
    const int NA = NH * NL * NP;      // 72
    const int NOpad = 192, NApad = 128;

    const float* x    = (const float*)d_in[0];
    const float* feat = (const float*)d_in[1];
    const float* refp = (const float*)d_in[2];
    const float* qn_g = (const float*)d_in[5];
    const float* qn_b = (const float*)d_in[6];
    const float* fn_g = (const float*)d_in[7];
    const float* fn_b = (const float*)d_in[8];
    const float* Wv   = (const float*)d_in[9];
    const float* bv   = (const float*)d_in[10];
    const float* Wo   = (const float*)d_in[11];
    const float* bo   = (const float*)d_in[12];
    const float* Wa   = (const float*)d_in[13];
    const float* ba   = (const float*)d_in[14];
    const float* Wat  = (const float*)d_in[15];
    const float* bat  = (const float*)d_in[16];
    const float* gam  = (const float*)d_in[17];
    const float* Wp   = (const float*)d_in[18];
    const float* bp   = (const float*)d_in[19];

    float* out_x      = (float*)d_out;                         // x_out [B,Q,C]
    float* out_prompt = out_x + (size_t)MQ * C;                // prompt [B,Q,C]

    // workspace layout
    char* w = (char*)d_ws;
    size_t cur = 0;
    auto take = [&](size_t bytes) -> char* {
        char* p = w + cur;
        cur += (bytes + 255) & ~(size_t)255;
        return p;
    };
    __bf16* q_bf      = (__bf16*)take((size_t)MQ * C * 2);
    __bf16* v_bf      = (__bf16*)take((size_t)MV * C * 2);
    float*  value_f   = (float*) take((size_t)MV * C * 4);
    float*  off_f     = (float*) take((size_t)MQ * NO * 4);
    float*  aw_f      = (float*) take((size_t)MQ * NA * 4);
    __bf16* attn_bf   = (__bf16*)take((size_t)MQ * C * 2);
    __bf16* prompt_bf = (__bf16*)take((size_t)MQ * C * 2);
    __bf16* wt_v      = (__bf16*)take((size_t)C * C * 2);
    __bf16* wt_o      = (__bf16*)take((size_t)NOpad * C * 2);
    __bf16* wt_a      = (__bf16*)take((size_t)NApad * C * 2);
    __bf16* wt_at     = (__bf16*)take((size_t)C * C * 2);
    __bf16* wt_p      = (__bf16*)take((size_t)C * C * 2);

    // 1) weight transposes (f32 -> bf16, [Npad][768])
    auto tgrid = [](int npad) { return dim3((npad * KDIM + 255) / 256); };
    wtrans_kernel<<<tgrid(C),     256, 0, stream>>>(Wv,  wt_v,  C,  C);
    wtrans_kernel<<<tgrid(NOpad), 256, 0, stream>>>(Wo,  wt_o,  NO, NOpad);
    wtrans_kernel<<<tgrid(NApad), 256, 0, stream>>>(Wa,  wt_a,  NA, NApad);
    wtrans_kernel<<<tgrid(C),     256, 0, stream>>>(Wat, wt_at, C,  C);
    wtrans_kernel<<<tgrid(C),     256, 0, stream>>>(Wp,  wt_p,  C,  C);

    // 2) layernorms -> bf16 activations
    ln_kernel<<<MQ, 256, 0, stream>>>(x,    qn_g, qn_b, q_bf);
    ln_kernel<<<MV, 256, 0, stream>>>(feat, fn_g, fn_b, v_bf);

    // 3) value = v_in @ Wv + bv -> head-major [B,H,Lin,dh] f32
    gemm_wmma_kernel<0><<<dim3(C / 64, MV / 128), 256, 0, stream>>>(
        v_bf, wt_v, bv, nullptr, nullptr, value_f, nullptr, MV, C);

    // 4) sampling offsets = q @ Wo + bo  [MQ,144]
    gemm_wmma_kernel<1><<<dim3(NOpad / 64, MQ / 128), 256, 0, stream>>>(
        q_bf, wt_o, bo, nullptr, nullptr, off_f, nullptr, MQ, NO);

    // 5) attention logits = q @ Wa + ba  [MQ,72]
    gemm_wmma_kernel<1><<<dim3(NApad / 64, MQ / 128), 256, 0, stream>>>(
        q_bf, wt_a, ba, nullptr, nullptr, aw_f, nullptr, MQ, NA);

    // 6) deformable sampling -> attn_out bf16 [MQ,768]
    msdeform_kernel<<<dim3(MQ * NH), 128, 0, stream>>>(
        off_f, aw_f, refp, value_f, attn_bf);

    // 7) prompt = (attn_out @ Wattn + battn) * gamma -> d_out(second) + bf16 copy
    gemm_wmma_kernel<2><<<dim3(C / 64, MQ / 128), 256, 0, stream>>>(
        attn_bf, wt_at, bat, gam, nullptr, out_prompt, prompt_bf, MQ, C);

    // 8) x_out = x + prompt @ Wp + bp -> d_out(first)
    gemm_wmma_kernel<3><<<dim3(C / 64, MQ / 128), 256, 0, stream>>>(
        prompt_bf, wt_p, bp, nullptr, x, out_x, nullptr, MQ, C);
}
